// RecModel_12807592477135
// MI455X (gfx1250) — matrix operand order repaired and checked by reading, not compiled, not run on hardware
//
#include <hip/hip_runtime.h>
#include <hip/hip_bf16.h>
#include <math.h>

typedef __attribute__((ext_vector_type(2)))  _Float16 v2h;
typedef __attribute__((ext_vector_type(8)))  _Float16 v8h;
typedef __attribute__((ext_vector_type(16))) _Float16 v16h;
typedef __attribute__((ext_vector_type(8)))  float    v8f;

#define BATCH   16384
#define NS      20
#define VOCAB   100000
#define EMB     16
#define DENSE   13
#define XCOLS   (NS + DENSE)        // 33
#define K1ACT   3373                // 20*16 + 190*16 + 13
#define K1PAD   3392                // padded to multiple of 32
#define LDA1    3400                // feature row stride (halfs), 16B-aligned
#define ED      128
#define NC      100
#define MROWS   32                  // batch rows per workgroup (2 WMMA M-tiles)

struct RecParams {
  const float *x, *tables;
  const float *b1,*g1,*be1,*m1,*v1;
  const float *b2,*g2,*be2,*m2,*v2;
  const float *b3;
  const float *bc;
  const float *bp1,*gp,*bep,*mp,*vp;
  const float *bp2;
  const _Float16 *W1t,*W2t,*W3t,*Wct,*Wp1t,*Wp2t;  // packed f16, [Npad][Kpad]
  float *logits, *proj;
};

// ---- pack: W (f32, row-major [Kact x Nact]) -> Wt (f16, [Npad x Kpad], zero pad) ----
__global__ __launch_bounds__(256) void pack_w_kernel(const float* __restrict__ W,
                                                     _Float16* __restrict__ Wt,
                                                     int Kact, int Kpad, int Nact,
                                                     int total)
{
  int i = blockIdx.x * 256 + threadIdx.x;
  if (i >= total) return;
  int n = i / Kpad, k = i - n * Kpad;
  float v = (n < Nact && k < Kact) ? W[(size_t)k * Nact + n] : 0.f;
  Wt[i] = (_Float16)v;
}

__device__ inline void store2(_Float16* p, float a, float b) {
  v2h t; t[0] = (_Float16)a; t[1] = (_Float16)b;
  *(v2h*)p = t;                       // one 4B LDS store
}

// MT 16-row tiles (LDS f16 A) @ packed f16 B (global) -> f32 acc[MT][NPAD/128].
// Each B-fragment (32 contiguous bytes/lane) is reused across MT M-tiles.
template<int NPAD, int MT>
__device__ inline void gemm_mt(const _Float16* __restrict__ sA, int lda,
                               const _Float16* __restrict__ Wt, int Kpad,
                               v8f acc[MT][NPAD / 128], int tid)
{
  constexpr int TPW = NPAD / 128;       // n-tiles per wave (8 waves)
  const int wave = tid >> 5, lane = tid & 31;
  const int m   = lane & 15;
  const int kb  = (lane >> 4) * 16;     // A-frag: lanes<16 K 0..15, lanes>=16 K 16..31
  const int kb2 = (lane >> 4) * 16;     // B-frag: lanes<16 K 0..15, lanes>=16 K 16..31
  const _Float16* __restrict__ arow[MT];
  #pragma unroll
  for (int mt = 0; mt < MT; ++mt) arow[mt] = sA + (mt * 16 + m) * lda + kb;
  const _Float16* __restrict__ bbase[TPW];
  #pragma unroll
  for (int t = 0; t < TPW; ++t) {
    int n = (wave * TPW + t) * 16 + (lane & 15);
    bbase[t] = Wt + (size_t)n * Kpad + kb2;
  }
  #pragma unroll 2
  for (int k0 = 0; k0 < Kpad; k0 += 32) {
    v16h a[MT];
    #pragma unroll
    for (int mt = 0; mt < MT; ++mt) {
      v8h alo = *(const v8h*)(arow[mt] + k0);
      v8h ahi = *(const v8h*)(arow[mt] + k0 + 8);
      a[mt] = __builtin_shufflevector(alo, ahi,
                0,1,2,3,4,5,6,7,8,9,10,11,12,13,14,15);
    }
    __builtin_prefetch(bbase[0] + k0 + 512, 0, 1);   // next W cachelines
    #pragma unroll
    for (int t = 0; t < TPW; ++t) {
      const _Float16* bp = bbase[t] + k0;
      v8h blo = *(const v8h*)bp;
      v8h bhi = *(const v8h*)(bp + 8);
      v16h b  = __builtin_shufflevector(blo, bhi,
                  0,1,2,3,4,5,6,7,8,9,10,11,12,13,14,15);
      #pragma unroll
      for (int mt = 0; mt < MT; ++mt)
        acc[mt][t] = __builtin_amdgcn_wmma_f32_16x16x32_f16(
            false, a[mt], false, b, (short)0, acc[mt][t], false, false);
    }
  }
}

// epilogue: bias (+BN) (+ReLU), f32 acc -> f16 LDS tile [MROWS][ldo]
template<int NPAD, int MT, bool HASBN, bool RELU>
__device__ inline void epi_to_lds(v8f acc[MT][NPAD / 128], const float* b, const float* g,
                                  const float* be, const float* mean, const float* var,
                                  _Float16* sOut, int ldo, int tid)
{
  constexpr int TPW = NPAD / 128;
  const int wave = tid >> 5, lane = tid & 31;
  #pragma unroll
  for (int t = 0; t < TPW; ++t) {
    int n = (wave * TPW + t) * 16 + (lane & 15);
    float bb = b[n];
    float sc = 1.f, sh = bb;
    if constexpr (HASBN) {
      float iv = rsqrtf(var[n] + 1e-5f);
      sc = g[n] * iv;
      sh = be[n] + (bb - mean[n]) * sc;
    }
    #pragma unroll
    for (int mt = 0; mt < MT; ++mt) {
      #pragma unroll
      for (int v = 0; v < 8; ++v) {
        int mm = mt * 16 + v + (lane >> 4) * 8;
        float xv = acc[mt][t][v] * sc + sh;
        if constexpr (RELU) xv = fmaxf(xv, 0.f);
        sOut[mm * ldo + n] = (_Float16)xv;
      }
    }
  }
}

__global__ __launch_bounds__(256) void rec_fused_kernel(RecParams P)
{
  // Feature tile: 32 rows x 3400 halfs = 217600 B; activation buffers overlay it
  // (features are dead after the L1 GEMM; barriers separate the phases).
  __shared__ __align__(16) unsigned char sMem[MROWS * LDA1 * 2];
  _Float16 (*sA)[LDA1] = (_Float16(*)[LDA1])sMem;
  _Float16 (*sH )[256] = (_Float16(*)[256])(sMem + 0);       // 16 KB
  _Float16 (*sH2)[256] = (_Float16(*)[256])(sMem + 16384);   // 16 KB
  _Float16 (*sZ )[ED]  = (_Float16(*)[ED]) (sMem + 32768);   // 8 KB
  _Float16 (*sPh)[ED]  = (_Float16(*)[ED]) (sMem + 40960);   // 8 KB
  float    (*sP )[ED]  = (float(*)[ED])    (sMem + 49152);   // 16 KB
  float    *sNorm      = (float*)          (sMem + 65536);

  const int tid = threadIdx.x;
  const int r0  = blockIdx.x * MROWS;

  // ---- build 32x3392 feature tile: [emb(320) | inter(3040) | dense(13) | pad] ----
  // 8 threads per row, each owns an even column pair -> packed b32 LDS stores.
  {
    const int row = tid >> 3;             // 0..31
    const int c0  = (tid & 7) * 2;        // 0,2,..,14
    const float* xr = P.x + (size_t)(r0 + row) * XCOLS;
    float ea[NS], eb[NS];
    #pragma unroll
    for (int s = 0; s < NS; ++s) {
      int idx = (int)xr[s];
      const float* te = P.tables + ((size_t)s * VOCAB + (size_t)idx) * EMB + c0;
      ea[s] = te[0]; eb[s] = te[1];
      store2(&sA[row][s * EMB + c0], ea[s], eb[s]);
    }
    int p = 0;
    #pragma unroll
    for (int i = 0; i < NS - 1; ++i)
      #pragma unroll
      for (int j = i + 1; j < NS; ++j) {
        store2(&sA[row][NS * EMB + p * EMB + c0], ea[i] * ea[j], eb[i] * eb[j]);
        ++p;
      }
    float d0 = (c0     < DENSE) ? xr[NS + c0]     : 0.f;
    float d1 = (c0 + 1 < DENSE) ? xr[NS + c0 + 1] : 0.f;
    store2(&sA[row][3360 + c0], d0, d1);   // dense 3360..3372, zeros beyond
    store2(&sA[row][3376 + c0], 0.f, 0.f);
  }
  __syncthreads();

  v8f acc2[2][2], acc1[2][1];

  // L1: (32x3392) @ W1t + bn1 + relu
  #pragma unroll
  for (int mt = 0; mt < 2; ++mt) for (int t = 0; t < 2; ++t)
    for (int i = 0; i < 8; ++i) acc2[mt][t][i] = 0.f;
  gemm_mt<256, 2>(&sA[0][0], LDA1, P.W1t, K1PAD, acc2, tid);
  __syncthreads();   // all waves done reading sA before overlay writes
  epi_to_lds<256, 2, true, true>(acc2, P.b1, P.g1, P.be1, P.m1, P.v1, &sH[0][0], 256, tid);
  __syncthreads();

  // L2: 256 -> 256 + bn2 + relu
  #pragma unroll
  for (int mt = 0; mt < 2; ++mt) for (int t = 0; t < 2; ++t)
    for (int i = 0; i < 8; ++i) acc2[mt][t][i] = 0.f;
  gemm_mt<256, 2>(&sH[0][0], 256, P.W2t, 256, acc2, tid);
  __syncthreads();
  epi_to_lds<256, 2, true, true>(acc2, P.b2, P.g2, P.be2, P.m2, P.v2, &sH2[0][0], 256, tid);
  __syncthreads();

  // L3: 256 -> 128 (bias only) => z
  #pragma unroll
  for (int mt = 0; mt < 2; ++mt) for (int i = 0; i < 8; ++i) acc1[mt][0][i] = 0.f;
  gemm_mt<128, 2>(&sH2[0][0], 256, P.W3t, 256, acc1, tid);
  __syncthreads();
  epi_to_lds<128, 2, false, false>(acc1, P.b3, nullptr, nullptr, nullptr, nullptr,
                                   &sZ[0][0], ED, tid);
  __syncthreads();

  // cls: z(128) -> logits(100), straight to global (Wct zero-padded to 128 cols)
  #pragma unroll
  for (int mt = 0; mt < 2; ++mt) for (int i = 0; i < 8; ++i) acc1[mt][0][i] = 0.f;
  gemm_mt<128, 2>(&sZ[0][0], ED, P.Wct, ED, acc1, tid);
  {
    const int wave = tid >> 5, lane = tid & 31;
    const int n = wave * 16 + (lane & 15);
    if (n < NC) {
      float bb = P.bc[n];
      #pragma unroll
      for (int mt = 0; mt < 2; ++mt)
        #pragma unroll
        for (int v = 0; v < 8; ++v) {
          int mm = mt * 16 + v + (lane >> 4) * 8;
          P.logits[(size_t)(r0 + mm) * NC + n] = acc1[mt][0][v] + bb;
        }
    }
  }

  // p1: z -> 128 + bnp + relu
  #pragma unroll
  for (int mt = 0; mt < 2; ++mt) for (int i = 0; i < 8; ++i) acc1[mt][0][i] = 0.f;
  gemm_mt<128, 2>(&sZ[0][0], ED, P.Wp1t, ED, acc1, tid);
  __syncthreads();
  epi_to_lds<128, 2, true, true>(acc1, P.bp1, P.gp, P.bep, P.mp, P.vp, &sPh[0][0], ED, tid);
  __syncthreads();

  // p2: 128 -> 128 (bias only), keep f32 for row normalization
  #pragma unroll
  for (int mt = 0; mt < 2; ++mt) for (int i = 0; i < 8; ++i) acc1[mt][0][i] = 0.f;
  gemm_mt<128, 2>(&sPh[0][0], ED, P.Wp2t, ED, acc1, tid);
  __syncthreads();
  {
    const int wave = tid >> 5, lane = tid & 31;
    const int n = wave * 16 + (lane & 15);
    float bb = P.bp2[n];
    #pragma unroll
    for (int mt = 0; mt < 2; ++mt)
      #pragma unroll
      for (int v = 0; v < 8; ++v) {
        int mm = mt * 16 + v + (lane >> 4) * 8;
        sP[mm][n] = acc1[mt][0][v] + bb;
      }
  }
  __syncthreads();
  if (tid < MROWS) {
    float s = 0.f;
    #pragma unroll 8
    for (int c = 0; c < ED; ++c) { float pv = sP[tid][c]; s += pv * pv; }
    sNorm[tid] = fmaxf(sqrtf(s), 1e-12f);
  }
  __syncthreads();
  for (int e = tid; e < MROWS * ED; e += 256) {
    int mm = e >> 7, n = e & 127;
    P.proj[(size_t)(r0 + mm) * ED + n] = sP[mm][n] / sNorm[mm];
  }
}

extern "C" void kernel_launch(void* const* d_in, const int* in_sizes, int n_in,
                              void* d_out, int out_size, void* d_ws, size_t ws_size,
                              hipStream_t stream)
{
  const float* W1  = (const float*)d_in[2];
  const float* W2  = (const float*)d_in[8];
  const float* W3  = (const float*)d_in[14];
  const float* Wc  = (const float*)d_in[16];
  const float* Wp1 = (const float*)d_in[18];
  const float* Wp2 = (const float*)d_in[24];

  // packed weight layouts in workspace (f16, [Npad][Kpad]) — ~2.03 MB total
  _Float16* ws   = (_Float16*)d_ws;
  _Float16* W1t  = ws;                       // 256*3392
  _Float16* W2t  = W1t + 256 * K1PAD;        // 256*256
  _Float16* W3t  = W2t + 256 * 256;          // 128*256
  _Float16* Wct  = W3t + 128 * 256;          // 128*128 (cols 100..127 zero)
  _Float16* Wp1t = Wct + 128 * 128;          // 128*128
  _Float16* Wp2t = Wp1t + 128 * 128;         // 128*128

  auto pack = [&](const float* W, _Float16* Wt, int Kact, int Kpad, int Nact, int Npad) {
    int total = Npad * Kpad;
    pack_w_kernel<<<(total + 255) / 256, 256, 0, stream>>>(W, Wt, Kact, Kpad, Nact, total);
  };
  pack(W1,  W1t,  K1ACT, K1PAD, 256, 256);
  pack(W2,  W2t,  256,   256,   256, 256);
  pack(W3,  W3t,  256,   256,   ED,  ED);
  pack(Wc,  Wct,  ED,    ED,    NC,  ED);
  pack(Wp1, Wp1t, ED,    ED,    ED,  ED);
  pack(Wp2, Wp2t, ED,    ED,    ED,  ED);

  RecParams P;
  P.x      = (const float*)d_in[0];
  P.tables = (const float*)d_in[1];
  P.b1  = (const float*)d_in[3];
  P.g1  = (const float*)d_in[4];  P.be1 = (const float*)d_in[5];
  P.m1  = (const float*)d_in[6];  P.v1  = (const float*)d_in[7];
  P.b2  = (const float*)d_in[9];
  P.g2  = (const float*)d_in[10]; P.be2 = (const float*)d_in[11];
  P.m2  = (const float*)d_in[12]; P.v2  = (const float*)d_in[13];
  P.b3  = (const float*)d_in[15];
  P.bc  = (const float*)d_in[17];
  P.bp1 = (const float*)d_in[19];
  P.gp  = (const float*)d_in[20]; P.bep = (const float*)d_in[21];
  P.mp  = (const float*)d_in[22]; P.vp  = (const float*)d_in[23];
  P.bp2 = (const float*)d_in[25];
  P.W1t = W1t; P.W2t = W2t; P.W3t = W3t; P.Wct = Wct; P.Wp1t = Wp1t; P.Wp2t = Wp2t;
  P.logits = (float*)d_out;
  P.proj   = (float*)d_out + (size_t)BATCH * NC;

  rec_fused_kernel<<<BATCH / MROWS, 256, 0, stream>>>(P);
}
